// RecurrentCellWrapper_52381421142740
// MI455X (gfx1250) — compile-verified, matrix operand order checked by reading, not run on hardware
//
#include <hip/hip_runtime.h>
#include <hip/hip_bf16.h>
#include <math.h>

// Problem constants (from the reference): B=16, S=2048, I=1024, H=1024.
// out[b,s,h] = tanh( x[b,s,:] . W_ih[h,:] + hterm[b,h] )
// hterm[b,h] = hx[b,:] . W_hh[h,:] + b_ih[h] + b_hh[h]
#define BB   16
#define SS   2048
#define II   1024
#define HH   1024
#define MM   (BB * SS)          // 32768 rows of the big GEMM
#define KT   32                 // K tile (bf16 WMMA K)
#define NIT  (II / KT)          // 32 K iterations

typedef __attribute__((ext_vector_type(16))) __bf16 v16bf;
typedef __attribute__((ext_vector_type(2)))  __bf16 v2bf;
typedef __attribute__((ext_vector_type(8)))  float  v8f;

// ---------------------------------------------------------------------------
// Kernel 1: hterm[b][h] = hx[b]·W_hh[h,:] + b_ih[h] + b_hh[h]   (tiny: 33 MFLOP)
// ---------------------------------------------------------------------------
__global__ __launch_bounds__(256) void hterm_kernel(const float* __restrict__ hx,
                                                    const float* __restrict__ Whh,
                                                    const float* __restrict__ b_ih,
                                                    const float* __restrict__ b_hh,
                                                    float* __restrict__ hterm) {
    const int idx = blockIdx.x * blockDim.x + threadIdx.x;   // 0 .. 16*1024-1
    const int b = idx >> 10;
    const int h = idx & 1023;
    const float* __restrict__ hr = hx + (size_t)b * HH;
    const float* __restrict__ wr = Whh + (size_t)h * HH;
    float s = 0.0f;
    #pragma unroll 4
    for (int j = 0; j < HH; j += 4) {
        float4 a = *(const float4*)(hr + j);
        float4 w = *(const float4*)(wr + j);
        s = fmaf(a.x, w.x, s);
        s = fmaf(a.y, w.y, s);
        s = fmaf(a.z, w.z, s);
        s = fmaf(a.w, w.w, s);
    }
    hterm[idx] = s + b_ih[h] + b_hh[h];
}

// ---------------------------------------------------------------------------
// fp32 x2 -> packed bf16x2.
//   1) hardware v_cvt_pk_bf16_f32 if the toolchain exposes it (1 op)
//   2) else: round-half-up + single v_perm_b32 byte-pack (3 ops/pair)
//   3) else: manual RNE bit twiddling
// ---------------------------------------------------------------------------
__device__ __forceinline__ unsigned int pk_bf16(float a, float b) {
#if __has_builtin(__builtin_amdgcn_cvt_pk_bf16_f32)
    union { v2bf v; unsigned int u; } c;
    c.v = __builtin_amdgcn_cvt_pk_bf16_f32(a, b);
    return c.u;
#elif __has_builtin(__builtin_amdgcn_perm)
    // Round-half-up at bit 16 (differs from RNE only on exact ties), then
    // v_perm_b32 packs {b[31:16], a[31:16]} in one instruction:
    // dst.b0=ra.b2, dst.b1=ra.b3 (sel bytes from src1), dst.b2=rb.b2,
    // dst.b3=rb.b3 (sel bytes 6,7 from src0) -> selector 0x07060302.
    unsigned int ra = __float_as_uint(a) + 0x8000u;
    unsigned int rb = __float_as_uint(b) + 0x8000u;
    return __builtin_amdgcn_perm(rb, ra, 0x07060302u);
#else
    unsigned int ua = __float_as_uint(a);
    unsigned int ub = __float_as_uint(b);
    ua = (ua + 0x7FFFu + ((ua >> 16) & 1u)) >> 16;
    ub = (ub + 0x7FFFu + ((ub >> 16) & 1u)) >> 16;
    return ua | (ub << 16);
#endif
}

// gfx1250 has a hardware v_tanh_f32 TRANS op; use it if the toolchain exposes it.
__device__ __forceinline__ float fast_tanh(float v) {
#if __has_builtin(__builtin_amdgcn_tanhf)
    return __builtin_amdgcn_tanhf(v);
#elif __has_builtin(__builtin_amdgcn_tanh_f32)
    return __builtin_amdgcn_tanh_f32(v);
#else
    return tanhf(v);
#endif
}

// ---------------------------------------------------------------------------
// Kernel 2: fused GEMM (bf16 WMMA, f32 acc) + hterm add + tanh
//   Block tile: 128 (M) x 128 (N), K step 32, double-buffered LDS.
//   256 threads = 8 wave32 waves in a 4(M) x 2(N) wave grid;
//   each wave: 32x64 sub-tile = 2x4 WMMA accumulators of 16x16.
// ---------------------------------------------------------------------------
#define LDSPITCH 40   // 32 K elems + 8 pad (keeps 16B alignment, kills bank conflicts)

__global__ __launch_bounds__(256, 2) void gemm_tanh_kernel(const float* __restrict__ x,
                                                           const float* __restrict__ Wih,
                                                           const float* __restrict__ hterm,
                                                           float* __restrict__ out) {
    __shared__ unsigned short As[2][128 * LDSPITCH];   // 2 x 10 KB, bf16 A tiles (x)
    __shared__ unsigned short Bs[2][128 * LDSPITCH];   // 2 x 10 KB, bf16 B tiles (W_ih)

    const int tid  = threadIdx.x;
    const int lane = tid & 31;
    const int wave = tid >> 5;
    const int wm   = wave & 3;      // wave M position (0..3)
    const int wn   = wave >> 2;     // wave N position (0..1)
    const int bm   = blockIdx.x;    // 0..255  (M tiles)
    const int bn   = blockIdx.y;    // 0..7    (N tiles)

    // staging role: 2 threads per row, each covers 16 consecutive K elements
    const int lrow = tid >> 1;           // 0..127
    const int lcol = (tid & 1) * 16;     // 0 or 16
    const int soff = lrow * LDSPITCH + lcol;
    const float* __restrict__ xrow = x   + (size_t)(bm * 128 + lrow) * II + lcol;
    const float* __restrict__ wrow = Wih + (size_t)(bn * 128 + lrow) * II + lcol;

    v8f acc[2][4];
    #pragma unroll
    for (int mi = 0; mi < 2; ++mi)
        #pragma unroll
        for (int ni = 0; ni < 4; ++ni)
            acc[mi][ni] = (v8f)(0.0f);

    union Frag { v16bf v; uint4 u[2]; };

    const int klo = (lane >> 4) * 8;    // per-lane K sub-chunk (ISA 16-bit layout)
    const int l15 = lane & 15;

    // ---- prologue: stage tile 0 into buffer 0 ----
    {
        float4 fx0 = *(const float4*)(xrow);
        float4 fx1 = *(const float4*)(xrow + 4);
        float4 fx2 = *(const float4*)(xrow + 8);
        float4 fx3 = *(const float4*)(xrow + 12);
        float4 fw0 = *(const float4*)(wrow);
        float4 fw1 = *(const float4*)(wrow + 4);
        float4 fw2 = *(const float4*)(wrow + 8);
        float4 fw3 = *(const float4*)(wrow + 12);
        uint4 ax0, ax1, aw0, aw1;
        ax0.x = pk_bf16(fx0.x, fx0.y);  ax0.y = pk_bf16(fx0.z, fx0.w);
        ax0.z = pk_bf16(fx1.x, fx1.y);  ax0.w = pk_bf16(fx1.z, fx1.w);
        ax1.x = pk_bf16(fx2.x, fx2.y);  ax1.y = pk_bf16(fx2.z, fx2.w);
        ax1.z = pk_bf16(fx3.x, fx3.y);  ax1.w = pk_bf16(fx3.z, fx3.w);
        aw0.x = pk_bf16(fw0.x, fw0.y);  aw0.y = pk_bf16(fw0.z, fw0.w);
        aw0.z = pk_bf16(fw1.x, fw1.y);  aw0.w = pk_bf16(fw1.z, fw1.w);
        aw1.x = pk_bf16(fw2.x, fw2.y);  aw1.y = pk_bf16(fw2.z, fw2.w);
        aw1.z = pk_bf16(fw3.x, fw3.y);  aw1.w = pk_bf16(fw3.z, fw3.w);
        *(uint4*)(As[0] + soff)     = ax0;
        *(uint4*)(As[0] + soff + 8) = ax1;
        *(uint4*)(Bs[0] + soff)     = aw0;
        *(uint4*)(Bs[0] + soff + 8) = aw1;
    }
    __syncthreads();

    // ---- main loop: one barrier per iteration (double buffer) ----
    for (int it = 0; it < NIT; ++it) {
        const int cur  = it & 1;
        const int nxt  = cur ^ 1;
        const bool hasNext = (it + 1) < NIT;

        // 1) issue global loads for tile it+1 (hide under the 8 WMMAs below)
        float4 fx0, fx1, fx2, fx3, fw0, fw1, fw2, fw3;
        if (hasNext) {
            const int k0 = (it + 1) * KT;
            if (k0 + KT < II) {   // prefetch tile it+2 into GL2
                __builtin_prefetch(xrow + k0 + KT, 0, 1);
                __builtin_prefetch(wrow + k0 + KT, 0, 1);
            }
            fx0 = *(const float4*)(xrow + k0);
            fx1 = *(const float4*)(xrow + k0 + 4);
            fx2 = *(const float4*)(xrow + k0 + 8);
            fx3 = *(const float4*)(xrow + k0 + 12);
            fw0 = *(const float4*)(wrow + k0);
            fw1 = *(const float4*)(wrow + k0 + 4);
            fw2 = *(const float4*)(wrow + k0 + 8);
            fw3 = *(const float4*)(wrow + k0 + 12);
        }

        // 2) fragments from current buffer (two ds_load_b128 each, ISA layout)
        Frag a[2], b[4];
        #pragma unroll
        for (int mi = 0; mi < 2; ++mi) {
            const unsigned short* p = As[cur] + (wm * 32 + mi * 16 + l15) * LDSPITCH + klo;
            a[mi].u[0] = *(const uint4*)(p);
            a[mi].u[1] = *(const uint4*)(p + 16);
        }
        #pragma unroll
        for (int ni = 0; ni < 4; ++ni) {
            const unsigned short* p = Bs[cur] + (wn * 64 + ni * 16 + l15) * LDSPITCH + klo;
            b[ni].u[0] = *(const uint4*)(p);
            b[ni].u[1] = *(const uint4*)(p + 16);
        }

        // 3) 8x v_wmma_f32_16x16x32_bf16
        #pragma unroll
        for (int mi = 0; mi < 2; ++mi)
            #pragma unroll
            for (int ni = 0; ni < 4; ++ni)
                acc[mi][ni] = __builtin_amdgcn_wmma_f32_16x16x32_bf16(
                    /*neg_a=*/false, a[mi].v, /*neg_b=*/false, b[ni].v,
                    /*c_mod=*/(short)0, acc[mi][ni],
                    /*reuse_a=*/false, /*reuse_b=*/false);

        // 4) convert + store tile it+1 into the other buffer, single barrier
        if (hasNext) {
            uint4 ax0, ax1, aw0, aw1;
            ax0.x = pk_bf16(fx0.x, fx0.y);  ax0.y = pk_bf16(fx0.z, fx0.w);
            ax0.z = pk_bf16(fx1.x, fx1.y);  ax0.w = pk_bf16(fx1.z, fx1.w);
            ax1.x = pk_bf16(fx2.x, fx2.y);  ax1.y = pk_bf16(fx2.z, fx2.w);
            ax1.z = pk_bf16(fx3.x, fx3.y);  ax1.w = pk_bf16(fx3.z, fx3.w);
            aw0.x = pk_bf16(fw0.x, fw0.y);  aw0.y = pk_bf16(fw0.z, fw0.w);
            aw0.z = pk_bf16(fw1.x, fw1.y);  aw0.w = pk_bf16(fw1.z, fw1.w);
            aw1.x = pk_bf16(fw2.x, fw2.y);  aw1.y = pk_bf16(fw2.z, fw2.w);
            aw1.z = pk_bf16(fw3.x, fw3.y);  aw1.w = pk_bf16(fw3.z, fw3.w);
            *(uint4*)(As[nxt] + soff)     = ax0;
            *(uint4*)(As[nxt] + soff + 8) = ax1;
            *(uint4*)(Bs[nxt] + soff)     = aw0;
            *(uint4*)(Bs[nxt] + soff + 8) = aw1;
            __syncthreads();
        }
    }

    // ---- epilogue: + hterm, tanh, store ----
    // Tile rows are 128-aligned and S=2048 is a multiple of 128 => one batch per block.
    const int bidx = bm >> 4;                       // (bm*128) / 2048
    float ht[4];
    #pragma unroll
    for (int ni = 0; ni < 4; ++ni) {
        const int col = bn * 128 + wn * 64 + ni * 16 + l15;
        ht[ni] = hterm[bidx * HH + col];
    }

    const int mrow0 = bm * 128 + wm * 32 + ((lane >> 4) * 8);  // lanes 16-31: M += 8
    #pragma unroll
    for (int mi = 0; mi < 2; ++mi) {
        #pragma unroll
        for (int ni = 0; ni < 4; ++ni) {
            const int col = bn * 128 + wn * 64 + ni * 16 + l15;
            #pragma unroll
            for (int r = 0; r < 8; ++r) {
                const int row = mrow0 + mi * 16 + r;           // VGPR r -> M = r (+8 for hi lanes)
                out[(size_t)row * HH + col] = fast_tanh(acc[mi][ni][r] + ht[ni]);
            }
        }
    }
}

// ---------------------------------------------------------------------------
// Launcher
// ---------------------------------------------------------------------------
extern "C" void kernel_launch(void* const* d_in, const int* in_sizes, int n_in,
                              void* d_out, int out_size, void* d_ws, size_t ws_size,
                              hipStream_t stream) {
    const float* x    = (const float*)d_in[0];   // [B,S,I]
    const float* hx   = (const float*)d_in[1];   // [B,H]
    const float* Wih  = (const float*)d_in[2];   // [H,I]
    const float* Whh  = (const float*)d_in[3];   // [H,H]
    const float* b_ih = (const float*)d_in[4];   // [H]
    const float* b_hh = (const float*)d_in[5];   // [H]
    float*       out  = (float*)d_out;           // [B,S,H]
    float*       hterm = (float*)d_ws;           // [B,H] = 64 KB scratch

    // 1) hterm (tiny)
    hterm_kernel<<<dim3((BB * HH) / 256), dim3(256), 0, stream>>>(hx, Whh, b_ih, b_hh, hterm);

    // 2) fused GEMM + bias + tanh
    dim3 grid(MM / 128, HH / 128);   // 256 x 8 blocks
    gemm_tanh_kernel<<<grid, dim3(256), 0, stream>>>(x, Wih, hterm, out);
}
